// AttentionInteractionBlockVN_55044300866299
// MI455X (gfx1250) — matrix operand order repaired and checked by reading, not compiled, not run on hardware
//
#include <hip/hip_runtime.h>
#include <hip/hip_bf16.h>

#define N_NODES 20000
#define N_EDGES 320000
#define PI_F 3.1415926535897932f

typedef __attribute__((ext_vector_type(16))) __bf16 v16bf;
typedef __attribute__((ext_vector_type(4)))  __bf16 bf16x4;
typedef __attribute__((ext_vector_type(8)))  float  v8f;

// ---------------- WMMA fragment helpers (CDNA5 wave32 layouts, ISA 7.12.2) ----------------

// A-matrix 16x32 bf16 fragment from LDS row-major [16][lda]:
// lane m=lane&15, hi=lane>>4; element j -> K = (j<8 ? j : j+8) + hi*8 (+k0)
// lowers to two ds_load_b128 per fragment.
__device__ __forceinline__ v16bf ldsA_frag(const __bf16* A, int lda, int k0, int lane) {
  const int m  = lane & 15;
  const int hi = (lane >> 4) & 1;
  const __bf16* p = A + m * lda + k0 + hi * 8;
  v16bf a;
#pragma unroll
  for (int j = 0; j < 8; ++j) { a[j] = p[j]; a[j + 8] = p[16 + j]; }
  return a;
}

// B-matrix 32x16 bf16 fragment: packed so each lane reads 16 contiguous bf16 (32B, coalesced)
__device__ __forceinline__ v16bf ldB_frag(const __bf16* Bpk, int lane) {
  return *(const v16bf*)(Bpk + (lane << 4));
}

// One 16x16 output tile, looping K in steps of 32.  numNT = N/16 tiles of packed B.
__device__ __forceinline__ v8f gemm_tile(const __bf16* A, int lda, const __bf16* Bpk,
                                         int nt, int numNT, int kSteps, int lane) {
  v8f acc = {};
  for (int kt = 0; kt < kSteps; ++kt) {
    v16bf a = ldsA_frag(A, lda, kt * 32, lane);
    v16bf b = ldB_frag(Bpk + (((size_t)kt * numNT + nt) << 9), lane);
    acc = __builtin_amdgcn_wmma_f32_16x16x32_bf16(false, a, false, b, (short)0, acc,
                                                  false, false);
  }
  return acc;
}

// Cooperative f32(LDS) -> bf16 A-staging, 4-wide: float4 in, bf16x4 (ds_store_b64) out.
// Requires w%4==0, c0%4==0, lda%4==0 (always true here).
__device__ __forceinline__ void stage_bf(const float* src, __bf16* dst, int lda, int c0,
                                         int w, int tid) {
  const int nq = (16 * w) >> 2;
  for (int i = tid; i < nq; i += 256) {
    int q = i << 2;
    int m = q / w, c = q - m * w;
    float4 f = *(const float4*)(src + q);
    bf16x4 b;
    b[0] = (__bf16)f.x; b[1] = (__bf16)f.y; b[2] = (__bf16)f.z; b[3] = (__bf16)f.w;
    *(bf16x4*)(dst + m * lda + c0 + c) = b;
  }
}

// Stage all three 16x64 spatial planes (contiguous [3][1024] f32) flat into bf16 LDS.
__device__ __forceinline__ void stage3_bf(const float* src, __bf16* dst, int tid) {
  for (int i = tid; i < 768; i += 256) {
    float4 f = ((const float4*)src)[i];
    bf16x4 b;
    b[0] = (__bf16)f.x; b[1] = (__bf16)f.y; b[2] = (__bf16)f.z; b[3] = (__bf16)f.w;
    ((bf16x4*)dst)[i] = b;
  }
}

__device__ __forceinline__ float fsqrt_fast(float x) { return __builtin_amdgcn_sqrtf(x); }

__device__ __forceinline__ float sigmoidf_(float x) { return 1.0f / (1.0f + __expf(-x)); }

__device__ __forceinline__ void atomAddF(float* p, float v) {
  __hip_atomic_fetch_add(p, v, __ATOMIC_RELAXED, __HIP_MEMORY_SCOPE_AGENT);
}

// ---------------- weight repack: f32 (OUT,IN) row-major -> packed bf16 WMMA B tiles ------

__global__ __launch_bounds__(256)
void pack_weight_kernel(const float* __restrict__ W, __bf16* __restrict__ dst,
                        int OUT, int IN) {
  int idx = blockIdx.x * 256 + threadIdx.x;
  if (idx >= OUT * IN) return;
  int tile = idx >> 9;           // 512 elems per (kt,nt) tile
  int lane = (idx >> 4) & 31;
  int j    = idx & 15;
  int numNT = OUT >> 4;
  int kt = tile / numNT, nt = tile - kt * numNT;
  int k = kt * 32 + ((lane >> 4) << 3) + ((j & 8) << 1) + (j & 7);     // B layout: K = (j<8 ? j : j+8) + hi*8
  int n = nt * 16 + (lane & 15);
  dst[idx] = (__bf16)W[(size_t)n * IN + k];     // B[K][N] = W[n][k]
}

__global__ __launch_bounds__(256)
void zero_kernel(float* p, long n) {
  long i = (long)blockIdx.x * 256 + threadIdx.x;
  long s = (long)gridDim.x * 256;
  for (; i < n; i += s) p[i] = 0.0f;
}

// ---------------- stage 0: node_gv + centroid GVLinear over all nodes --------------------

__global__ __launch_bounds__(256)
void node_gv_kernel(const float* __restrict__ xs, const float* __restrict__ xv,
                    const __bf16* nWv1, const __bf16* nWv2, const __bf16* nWg,
                    const float* nGb, const __bf16* nWs,
                    const __bf16* cWv1, const __bf16* cWv2, const __bf16* cWg,
                    const float* cGb, const __bf16* cWs,
                    float* __restrict__ ns, float* __restrict__ nv,
                    float* __restrict__ cs, float* __restrict__ cv) {
  __shared__ __align__(16) float  sS[16 * 256];     // x_sca (persistent)
  __shared__ __align__(16) float  sV[3][16 * 64];   // x_vec planes (persistent)
  __shared__ __align__(16) float  sVi[3][16 * 64];  // vi
  __shared__ __align__(16) float  sVn[16 * 64];     // vn, then gate
  __shared__ __align__(16) __bf16 sA[16 * 320];     // A staging
  __shared__ __align__(16) __bf16 sA2[16 * 256];    // out_s staging for gate GEMM

  const int tid = threadIdx.x, lane = tid & 31, wid = tid >> 5;
  const int n0 = blockIdx.x * 16;

  for (int i = tid; i < 1024; i += 256)
    ((float4*)sS)[i] = ((const float4*)(xs + (size_t)n0 * 256))[i];
  for (int i = tid; i < 3 * 16 * 64; i += 256) {
    int m = i / 192, rem = i - m * 192, c = rem / 3, k = rem - c * 3;
    sV[k][m * 64 + c] = xv[(size_t)(n0 + m) * 192 + rem];
  }
  __syncthreads();

  for (int pass = 0; pass < 2; ++pass) {
    const __bf16* W1 = pass ? cWv1 : nWv1;
    const __bf16* W2 = pass ? cWv2 : nWv2;
    const __bf16* Wg = pass ? cWg  : nWg;
    const float*  bg = pass ? cGb  : nGb;
    const __bf16* Wp = pass ? cWs  : nWs;
    float* outS = pass ? cs : ns;
    float* outV = pass ? cv : nv;

    stage3_bf(&sV[0][0], sA, tid); __syncthreads();  // vi = Wv1 @ xv (all 3 planes)
    for (int t = wid; t < 12; t += 8) {
      int k = t >> 2, nt = t & 3;
      v8f acc = gemm_tile(sA + k * 1024, 64, W1, nt, 4, 2, lane);
      int hi = lane >> 4, n = (lane & 15) + nt * 16;
#pragma unroll
      for (int r = 0; r < 8; ++r) sVi[k][(hi * 8 + r) * 64 + n] = acc[r];
    }
    __syncthreads();
    for (int i = tid; i < 1024; i += 256) {          // vn
      float a = sVi[0][i], b = sVi[1][i], c = sVi[2][i];
      sVn[i] = fsqrt_fast(a * a + b * b + c * c);
    }
    __syncthreads();
    stage_bf(sVn, sA, 320, 0, 64, tid);              // A = [vn | sca]
    stage_bf(sS,  sA, 320, 64, 256, tid);
    __syncthreads();
    for (int nt = wid; nt < 16; nt += 8) {           // out_s = A @ Ws^T
      v8f acc = gemm_tile(sA, 320, Wp, nt, 16, 10, lane);
      int hi = lane >> 4, n = (lane & 15) + nt * 16;
#pragma unroll
      for (int r = 0; r < 8; ++r) {
        int m = hi * 8 + r;
        float v = acc[r];
        outS[(size_t)(n0 + m) * 256 + n] = v;
        sA2[m * 256 + n] = (__bf16)v;
      }
    }
    __syncthreads();
    for (int nt = wid; nt < 4; nt += 8) {            // gate = sigmoid(out_s @ Wg^T + b)
      v8f acc = gemm_tile(sA2, 256, Wg, nt, 4, 8, lane);
      int hi = lane >> 4, n = (lane & 15) + nt * 16;
#pragma unroll
      for (int r = 0; r < 8; ++r)
        sVn[(hi * 8 + r) * 64 + n] = sigmoidf_(acc[r] + bg[n]);
    }
    __syncthreads();
    stage3_bf(&sVi[0][0], sA, tid); __syncthreads(); // out_v = gate * (Wv2 @ vi)
    for (int t = wid; t < 12; t += 8) {
      int k = t >> 2, nt = t & 3;
      v8f acc = gemm_tile(sA + k * 1024, 64, W2, nt, 4, 2, lane);
      int hi = lane >> 4, n = (lane & 15) + nt * 16;
#pragma unroll
      for (int r = 0; r < 8; ++r) {
        int m = hi * 8 + r;
        outV[(size_t)(n0 + m) * 192 + k * 64 + n] = sVn[m * 64 + n] * acc[r];
      }
    }
    __syncthreads();
  }
}

// ---------------- stage 1: fully fused per-edge message + scatter ------------------------

struct EdgeArgs {
  const int*   eidx;
  const float* efeat;
  const float* evec;
  const float* vee_w;
  const __bf16 *eWv1, *eWv2, *eWg, *eWs;
  const float* eGb;
  const __bf16* eDir;
  const __bf16* scaW; const float* scaB;
  const __bf16* e2nW; const float* e2nB;
  const __bf16* n2eW; const float* n2eB;
  const __bf16* WevW;
  const __bf16 *oWv1, *oWv2, *oWg, *oWs;
  const float* oGb;
  const float* ns; const float* nv;
  float* aggS; float* aggV;
};

__global__ __launch_bounds__(256)
void edge_message_kernel(EdgeArgs p) {
  __shared__ __align__(16) float  F256[16 * 256];   // ns_g -> y_sca -> ms
  __shared__ __align__(16) float  F64a[16 * 64];    // edge_sca -> es -> a_e2n -> vn_o
  __shared__ __align__(16) float  F64b[16 * 64];    // vn_e -> gate_e -> b_n2e -> gate_o
  __shared__ __align__(16) float  V3a[3][16 * 64];  // edge_vec -> ev -> vi_o
  __shared__ __align__(16) float  V3b[3][16 * 64];  // vi_e -> dmat -> nv_g -> y_vec
  __shared__ __align__(16) __bf16 sA[16 * 320];
  __shared__ float  sD[16], sC[16];
  __shared__ int    sRow[16], sCol[16];

  const int tid = threadIdx.x, lane = tid & 31, wid = tid >> 5;
  const int e0 = blockIdx.x * 16;

  if (tid < 16) {
    int e = e0 + tid;
    float vx = p.evec[e * 3 + 0], vy = p.evec[e * 3 + 1], vz = p.evec[e * 3 + 2];
    float d = fsqrt_fast(vx * vx + vy * vy + vz * vz);
    sD[tid] = d;
    float cc = 0.5f * (__cosf(d * (PI_F / 10.0f)) + 1.0f);
    sC[tid] = (d <= 10.0f) ? cc : 0.0f;
    sRow[tid] = p.eidx[e];
    int col = p.eidx[N_EDGES + e];
    sCol[tid] = col;
#pragma unroll
    for (int q = 0; q < 8; ++q) __builtin_prefetch(p.ns + (size_t)col * 256 + q * 32, 0, 0);
#pragma unroll
    for (int q = 0; q < 6; ++q) __builtin_prefetch(p.nv + (size_t)col * 192 + q * 32, 0, 0);
  }
  __syncthreads();

  // edge_sca = [gaussian(d) | edge_feature], edge_vec = vee_w[c] * unit(v)
  const float gstep = 10.0f / 59.0f;
  const float gcoef = -0.5f / (gstep * gstep);
  for (int i = tid; i < 16 * 64; i += 256) {
    int m = i >> 6, c = i & 63;
    float v;
    if (c < 60) { float t = sD[m] - gstep * (float)c; v = __expf(gcoef * t * t); }
    else        v = p.efeat[(size_t)(e0 + m) * 4 + (c - 60)];
    F64a[i] = v;
  }
  for (int i = tid; i < 3 * 16 * 64; i += 256) {
    int k = i >> 10, r = i & 1023, m = r >> 6, c = r & 63;
    float u = p.evec[(size_t)(e0 + m) * 3 + k] / (sD[m] + 1e-7f);
    V3a[k][r] = p.vee_w[c] * u;
  }
  __syncthreads();

  // --- edge GVLinear: vi_e = eWv1 @ edge_vec (12 tiles, all planes) ---
  stage3_bf(&V3a[0][0], sA, tid); __syncthreads();
  for (int t = wid; t < 12; t += 8) {
    int k = t >> 2, nt = t & 3;
    v8f acc = gemm_tile(sA + k * 1024, 64, p.eWv1, nt, 4, 2, lane);
    int hi = lane >> 4, n = (lane & 15) + nt * 16;
#pragma unroll
    for (int r = 0; r < 8; ++r) V3b[k][(hi * 8 + r) * 64 + n] = acc[r];
  }
  __syncthreads();
  for (int i = tid; i < 1024; i += 256) {
    float a = V3b[0][i], b = V3b[1][i], c = V3b[2][i];
    F64b[i] = fsqrt_fast(a * a + b * b + c * c);
  }
  __syncthreads();
  stage_bf(F64b, sA, 128, 0, 64, tid);
  stage_bf(F64a, sA, 128, 64, 64, tid);
  __syncthreads();
  for (int nt = wid; nt < 4; nt += 8) {               // es_out = [vn|sca] @ eWs^T
    v8f acc = gemm_tile(sA, 128, p.eWs, nt, 4, 4, lane);
    int hi = lane >> 4, n = (lane & 15) + nt * 16;
#pragma unroll
    for (int r = 0; r < 8; ++r) F64a[(hi * 8 + r) * 64 + n] = acc[r];
  }
  __syncthreads();
  stage_bf(F64a, sA, 64, 0, 64, tid); __syncthreads();
  for (int nt = wid; nt < 4; nt += 8) {               // gate_e
    v8f acc = gemm_tile(sA, 64, p.eWg, nt, 4, 2, lane);
    int hi = lane >> 4, n = (lane & 15) + nt * 16;
#pragma unroll
    for (int r = 0; r < 8; ++r)
      F64b[(hi * 8 + r) * 64 + n] = sigmoidf_(acc[r] + p.eGb[n]);
  }
  __syncthreads();
  stage3_bf(&V3b[0][0], sA, tid); __syncthreads();    // ev = gate_e * (eWv2 @ vi_e)
  for (int t = wid; t < 12; t += 8) {
    int k = t >> 2, nt = t & 3;
    v8f acc = gemm_tile(sA + k * 1024, 64, p.eWv2, nt, 4, 2, lane);
    int hi = lane >> 4, n = (lane & 15) + nt * 16;
#pragma unroll
    for (int r = 0; r < 8; ++r) {
      int idx = (hi * 8 + r) * 64 + n;
      V3a[k][idx] = F64b[idx] * acc[r];
    }
  }
  __syncthreads();
  stage3_bf(&V3a[0][0], sA, tid); __syncthreads();    // dmat = edge_dir @ ev
  for (int t = wid; t < 12; t += 8) {
    int k = t >> 2, nt = t & 3;
    v8f acc = gemm_tile(sA + k * 1024, 64, p.eDir, nt, 4, 2, lane);
    int hi = lane >> 4, n = (lane & 15) + nt * 16;
#pragma unroll
    for (int r = 0; r < 8; ++r) V3b[k][(hi * 8 + r) * 64 + n] = acc[r];
  }
  __syncthreads();
  for (int i = tid; i < 1024; i += 256) {             // VN-leaky + es leaky
    float x0 = V3a[0][i], x1 = V3a[1][i], x2 = V3a[2][i];
    float d0 = V3b[0][i], d1 = V3b[1][i], d2 = V3b[2][i];
    float dot = x0 * d0 + x1 * d1 + x2 * d2;
    if (dot < 0.0f) {
      float s = 0.8f * dot / (d0 * d0 + d1 * d1 + d2 * d2 + 1e-6f);
      V3a[0][i] = x0 - s * d0; V3a[1][i] = x1 - s * d1; V3a[2][i] = x2 - s * d2;
    }
    float e = F64a[i];
    F64a[i] = (e >= 0.0f) ? e : 0.01f * e;
  }
  __syncthreads();

  // --- gather source node features ---
  for (int i = tid; i < 16 * 64; i += 256) {
    int m = i >> 6, q = i & 63;
    ((float4*)F256)[i] = ((const float4*)(p.ns + (size_t)sCol[m] * 256))[q];
  }
  for (int i = tid; i < 16 * 48; i += 256) {
    int m = i / 48, r = i - m * 48, k = r >> 4, q = r & 15;
    *(float4*)(&V3b[k][m * 64 + q * 4]) =
        *(const float4*)(p.nv + (size_t)sCol[m] * 192 + k * 64 + q * 4);
  }
  __syncthreads();

  // --- message combine ---
  stage_bf(F256, sA, 256, 0, 256, tid); __syncthreads();
  for (int nt = wid; nt < 4; nt += 8) {               // b_n2e = ns_g @ n2e^T + b
    v8f acc = gemm_tile(sA, 256, p.n2eW, nt, 4, 8, lane);
    int hi = lane >> 4, n = (lane & 15) + nt * 16;
#pragma unroll
    for (int r = 0; r < 8; ++r) F64b[(hi * 8 + r) * 64 + n] = acc[r] + p.n2eB[n];
  }
  __syncthreads();
  stage_bf(F64a, sA, 64, 0, 64, tid); __syncthreads();
  for (int t = wid; t < 20; t += 8) {
    if (t < 16) {                                     // y_sca = ns_g * (es @ sca_lin + b)
      v8f acc = gemm_tile(sA, 64, p.scaW, t, 16, 2, lane);
      int hi = lane >> 4, n = (lane & 15) + t * 16;
#pragma unroll
      for (int r = 0; r < 8; ++r) {
        int idx = (hi * 8 + r) * 256 + n;
        F256[idx] = F256[idx] * (acc[r] + p.scaB[n]);
      }
    } else {                                          // a_e2n = es @ e2n + b
      v8f acc = gemm_tile(sA, 64, p.e2nW, t - 16, 4, 2, lane);
      int hi = lane >> 4, n = (lane & 15) + (t - 16) * 16;
#pragma unroll
      for (int r = 0; r < 8; ++r) F64a[(hi * 8 + r) * 64 + n] = acc[r] + p.e2nB[n];
    }
  }
  __syncthreads();
  stage3_bf(&V3a[0][0], sA, tid); __syncthreads();    // y_vec = a*nv_g + b*(Wev@ev)
  for (int t = wid; t < 12; t += 8) {
    int k = t >> 2, nt = t & 3;
    v8f acc = gemm_tile(sA + k * 1024, 64, p.WevW, nt, 4, 2, lane);
    int hi = lane >> 4, n = (lane & 15) + nt * 16;
#pragma unroll
    for (int r = 0; r < 8; ++r) {
      int idx = (hi * 8 + r) * 64 + n;
      V3b[k][idx] = F64a[idx] * V3b[k][idx] + F64b[idx] * acc[r];
    }
  }
  __syncthreads();

  // --- out GVLinear ---
  stage3_bf(&V3b[0][0], sA, tid); __syncthreads();    // vi_o
  for (int t = wid; t < 12; t += 8) {
    int k = t >> 2, nt = t & 3;
    v8f acc = gemm_tile(sA + k * 1024, 64, p.oWv1, nt, 4, 2, lane);
    int hi = lane >> 4, n = (lane & 15) + nt * 16;
#pragma unroll
    for (int r = 0; r < 8; ++r) V3a[k][(hi * 8 + r) * 64 + n] = acc[r];
  }
  __syncthreads();
  for (int i = tid; i < 1024; i += 256) {
    float a = V3a[0][i], b = V3a[1][i], c = V3a[2][i];
    F64a[i] = fsqrt_fast(a * a + b * b + c * c);
  }
  __syncthreads();
  stage_bf(F64a, sA, 320, 0, 64, tid);
  stage_bf(F256, sA, 320, 64, 256, tid);
  __syncthreads();
  for (int nt = wid; nt < 16; nt += 8) {              // ms
    v8f acc = gemm_tile(sA, 320, p.oWs, nt, 16, 10, lane);
    int hi = lane >> 4, n = (lane & 15) + nt * 16;
#pragma unroll
    for (int r = 0; r < 8; ++r) F256[(hi * 8 + r) * 256 + n] = acc[r];
  }
  __syncthreads();
  stage_bf(F256, sA, 256, 0, 256, tid); __syncthreads();
  for (int nt = wid; nt < 4; nt += 8) {               // gate_o
    v8f acc = gemm_tile(sA, 256, p.oWg, nt, 4, 8, lane);
    int hi = lane >> 4, n = (lane & 15) + nt * 16;
#pragma unroll
    for (int r = 0; r < 8; ++r)
      F64b[(hi * 8 + r) * 64 + n] = sigmoidf_(acc[r] + p.oGb[n]);
  }
  __syncthreads();
  stage3_bf(&V3a[0][0], sA, tid); __syncthreads();    // mv scatter
  for (int t = wid; t < 12; t += 8) {
    int k = t >> 2, nt = t & 3;
    v8f acc = gemm_tile(sA + k * 1024, 64, p.oWv2, nt, 4, 2, lane);
    int hi = lane >> 4, n = (lane & 15) + nt * 16;
#pragma unroll
    for (int r = 0; r < 8; ++r) {
      int m = hi * 8 + r;
      atomAddF(&p.aggV[(size_t)sRow[m] * 192 + k * 64 + n],
               sC[m] * F64b[m * 64 + n] * acc[r]);
    }
  }
  for (int i = tid; i < 16 * 256; i += 256) {         // ms scatter
    int m = i >> 8, c = i & 255;
    atomAddF(&p.aggS[(size_t)sRow[m] * 256 + c], sC[m] * F256[i]);
  }
}

// ---------------- stage 2: LN + activations + out_tr GVLinear ---------------------------

struct OutArgs {
  const float* cs; const float* cv;
  const float* aggS; const float* aggV;
  const float* lnsg; const float* lnsb;
  const float* lnvg; const float* lnvb;
  const __bf16* aDir;
  const __bf16 *tWv1, *tWv2, *tWg, *tWs;
  const float* tGb;
  float* out;
};

__global__ __launch_bounds__(256)
void node_out_kernel(OutArgs p) {
  __shared__ __align__(16) float  F256[16 * 256];
  __shared__ __align__(16) float  F64a[16 * 64];
  __shared__ __align__(16) float  F64b[16 * 64];
  __shared__ __align__(16) float  V3a[3][16 * 64];
  __shared__ __align__(16) float  V3b[3][16 * 64];
  __shared__ __align__(16) __bf16 sA[16 * 320];
  __shared__ float  sMu[16], sIv[16];

  const int tid = threadIdx.x, lane = tid & 31, wid = tid >> 5;
  const int n0 = blockIdx.x * 16;

  for (int i = tid; i < 1024; i += 256) {
    float4 a = ((const float4*)(p.cs + (size_t)n0 * 256))[i];
    float4 b = ((const float4*)(p.aggS + (size_t)n0 * 256))[i];
    float4 r; r.x = a.x + b.x; r.y = a.y + b.y; r.z = a.z + b.z; r.w = a.w + b.w;
    ((float4*)F256)[i] = r;
  }
  for (int i = tid; i < 16 * 48; i += 256) {
    int m = i / 48, r = i - m * 48, k = r >> 4, q = r & 15;
    size_t g = (size_t)(n0 + m) * 192 + k * 64 + q * 4;
    float4 a = *(const float4*)(p.cv + g);
    float4 b = *(const float4*)(p.aggV + g);
    float4 s; s.x = a.x + b.x; s.y = a.y + b.y; s.z = a.z + b.z; s.w = a.w + b.w;
    *(float4*)(&V3a[k][m * 64 + q * 4]) = s;
  }
  __syncthreads();
  if (tid < 16) {
    float mu = 0.f;
    for (int c = 0; c < 256; ++c) mu += F256[tid * 256 + c];
    mu *= (1.0f / 256.0f);
    float va = 0.f;
    for (int c = 0; c < 256; ++c) { float t = F256[tid * 256 + c] - mu; va += t * t; }
    sMu[tid] = mu; sIv[tid] = rsqrtf(va * (1.0f / 256.0f) + 1e-5f);
  }
  __syncthreads();
  for (int i = tid; i < 16 * 256; i += 256) {
    int m = i >> 8, c = i & 255;
    float x = (F256[i] - sMu[m]) * sIv[m] * p.lnsg[c] + p.lnsb[c];
    F256[i] = (x >= 0.f) ? x : 0.01f * x;
  }
  __syncthreads();
  if (tid < 16) {
    float mu = 0.f;
    for (int k = 0; k < 3; ++k)
      for (int c = 0; c < 64; ++c) mu += V3a[k][tid * 64 + c];
    mu *= (1.0f / 192.0f);
    float va = 0.f;
    for (int k = 0; k < 3; ++k)
      for (int c = 0; c < 64; ++c) { float t = V3a[k][tid * 64 + c] - mu; va += t * t; }
    sMu[tid] = mu; sIv[tid] = rsqrtf(va * (1.0f / 192.0f) + 1e-5f);
  }
  __syncthreads();
  for (int i = tid; i < 3 * 16 * 64; i += 256) {
    int k = i >> 10, r = i & 1023, m = r >> 6, c = r & 63;
    V3a[k][r] = (V3a[k][r] - sMu[m]) * sIv[m] * p.lnvg[c * 3 + k] + p.lnvb[c * 3 + k];
  }
  __syncthreads();
  stage3_bf(&V3a[0][0], sA, tid); __syncthreads();    // act_dir VN-leaky
  for (int t = wid; t < 12; t += 8) {
    int k = t >> 2, nt = t & 3;
    v8f acc = gemm_tile(sA + k * 1024, 64, p.aDir, nt, 4, 2, lane);
    int hi = lane >> 4, n = (lane & 15) + nt * 16;
#pragma unroll
    for (int r = 0; r < 8; ++r) V3b[k][(hi * 8 + r) * 64 + n] = acc[r];
  }
  __syncthreads();
  for (int i = tid; i < 1024; i += 256) {
    float x0 = V3a[0][i], x1 = V3a[1][i], x2 = V3a[2][i];
    float d0 = V3b[0][i], d1 = V3b[1][i], d2 = V3b[2][i];
    float dot = x0 * d0 + x1 * d1 + x2 * d2;
    if (dot < 0.0f) {
      float s = 0.8f * dot / (d0 * d0 + d1 * d1 + d2 * d2 + 1e-6f);
      V3a[0][i] = x0 - s * d0; V3a[1][i] = x1 - s * d1; V3a[2][i] = x2 - s * d2;
    }
  }
  __syncthreads();
  stage3_bf(&V3a[0][0], sA, tid); __syncthreads();    // out_tr: vi
  for (int t = wid; t < 12; t += 8) {
    int k = t >> 2, nt = t & 3;
    v8f acc = gemm_tile(sA + k * 1024, 64, p.tWv1, nt, 4, 2, lane);
    int hi = lane >> 4, n = (lane & 15) + nt * 16;
#pragma unroll
    for (int r = 0; r < 8; ++r) V3b[k][(hi * 8 + r) * 64 + n] = acc[r];
  }
  __syncthreads();
  for (int i = tid; i < 1024; i += 256) {
    float a = V3b[0][i], b = V3b[1][i], c = V3b[2][i];
    F64a[i] = fsqrt_fast(a * a + b * b + c * c);
  }
  __syncthreads();
  stage_bf(F64a, sA, 320, 0, 64, tid);
  stage_bf(F256, sA, 320, 64, 256, tid);
  __syncthreads();
  for (int nt = wid; nt < 16; nt += 8) {              // final out_s
    v8f acc = gemm_tile(sA, 320, p.tWs, nt, 16, 10, lane);
    int hi = lane >> 4, n = (lane & 15) + nt * 16;
#pragma unroll
    for (int r = 0; r < 8; ++r) {
      int m = hi * 8 + r;
      float v = acc[r];
      p.out[(size_t)(n0 + m) * 256 + n] = v;
      F256[m * 256 + n] = v;
    }
  }
  __syncthreads();
  stage_bf(F256, sA, 256, 0, 256, tid); __syncthreads();
  for (int nt = wid; nt < 4; nt += 8) {               // gate
    v8f acc = gemm_tile(sA, 256, p.tWg, nt, 4, 8, lane);
    int hi = lane >> 4, n = (lane & 15) + nt * 16;
#pragma unroll
    for (int r = 0; r < 8; ++r)
      F64b[(hi * 8 + r) * 64 + n] = sigmoidf_(acc[r] + p.tGb[n]);
  }
  __syncthreads();
  const size_t vbase = (size_t)N_NODES * 256;
  stage3_bf(&V3b[0][0], sA, tid); __syncthreads();    // final out_v (N,64,3 interleaved)
  for (int t = wid; t < 12; t += 8) {
    int k = t >> 2, nt = t & 3;
    v8f acc = gemm_tile(sA + k * 1024, 64, p.tWv2, nt, 4, 2, lane);
    int hi = lane >> 4, n = (lane & 15) + nt * 16;
#pragma unroll
    for (int r = 0; r < 8; ++r) {
      int m = hi * 8 + r;
      p.out[vbase + (size_t)(n0 + m) * 192 + (size_t)n * 3 + k] =
          F64b[m * 64 + n] * acc[r];
    }
  }
}

// ---------------- host ------------------------------------------------------------------

extern "C" void kernel_launch(void* const* d_in, const int* in_sizes, int n_in,
                              void* d_out, int out_size, void* d_ws, size_t ws_size,
                              hipStream_t stream) {
  (void)in_sizes; (void)n_in; (void)out_size; (void)ws_size;

  const float* x_sca = (const float*)d_in[0];
  const float* x_vec = (const float*)d_in[1];
  const int*   eidx  = (const int*)d_in[2];
  const float* efeat = (const float*)d_in[3];
  const float* evec  = (const float*)d_in[4];
  const float* vee_w = (const float*)d_in[5];
  const float* nWv1 = (const float*)d_in[6];
  const float* nWv2 = (const float*)d_in[7];
  const float* nWg  = (const float*)d_in[8];
  const float* nGb  = (const float*)d_in[9];
  const float* nWs  = (const float*)d_in[10];
  const float* egWv1 = (const float*)d_in[11];
  const float* egWv2 = (const float*)d_in[12];
  const float* egWg  = (const float*)d_in[13];
  const float* egGb  = (const float*)d_in[14];
  const float* egWs  = (const float*)d_in[15];
  const float* eDir  = (const float*)d_in[16];
  const float* scaW  = (const float*)d_in[17];
  const float* scaB  = (const float*)d_in[18];
  const float* e2nW  = (const float*)d_in[19];
  const float* e2nB  = (const float*)d_in[20];
  const float* n2eW  = (const float*)d_in[21];
  const float* n2eB  = (const float*)d_in[22];
  const float* WevW  = (const float*)d_in[23];
  const float* oWv1 = (const float*)d_in[24];
  const float* oWv2 = (const float*)d_in[25];
  const float* oWg  = (const float*)d_in[26];
  const float* oGb  = (const float*)d_in[27];
  const float* oWs  = (const float*)d_in[28];
  const float* cWv1 = (const float*)d_in[29];
  const float* cWv2 = (const float*)d_in[30];
  const float* cWg  = (const float*)d_in[31];
  const float* cGb  = (const float*)d_in[32];
  const float* cWs  = (const float*)d_in[33];
  const float* aDir = (const float*)d_in[34];
  const float* tWv1 = (const float*)d_in[35];
  const float* tWv2 = (const float*)d_in[36];
  const float* tWg  = (const float*)d_in[37];
  const float* tGb  = (const float*)d_in[38];
  const float* tWs  = (const float*)d_in[39];
  const float* lnsg = (const float*)d_in[40];
  const float* lnsb = (const float*)d_in[41];
  const float* lnvg = (const float*)d_in[42];
  const float* lnvb = (const float*)d_in[43];

  char* cur = (char*)d_ws;
  auto alloc_bytes = [&](size_t b) -> void* {
    void* q = cur; cur += (b + 255) & ~(size_t)255; return q;
  };
  auto pack = [&](const float* W, int OUT, int IN) -> const __bf16* {
    __bf16* dst = (__bf16*)alloc_bytes((size_t)OUT * IN * 2);
    int total = OUT * IN;
    pack_weight_kernel<<<(total + 255) / 256, 256, 0, stream>>>(W, dst, OUT, IN);
    return dst;
  };

  const __bf16* pk_nWv1 = pack(nWv1, 64, 64);
  const __bf16* pk_nWv2 = pack(nWv2, 64, 64);
  const __bf16* pk_nWg  = pack(nWg, 64, 256);
  const __bf16* pk_nWs  = pack(nWs, 256, 320);
  const __bf16* pk_eWv1 = pack(egWv1, 64, 64);
  const __bf16* pk_eWv2 = pack(egWv2, 64, 64);
  const __bf16* pk_eWg  = pack(egWg, 64, 64);
  const __bf16* pk_eWs  = pack(egWs, 64, 128);
  const __bf16* pk_eDir = pack(eDir, 64, 64);
  const __bf16* pk_scaW = pack(scaW, 256, 64);
  const __bf16* pk_e2nW = pack(e2nW, 64, 64);
  const __bf16* pk_n2eW = pack(n2eW, 64, 256);
  const __bf16* pk_WevW = pack(WevW, 64, 64);
  const __bf16* pk_oWv1 = pack(oWv1, 64, 64);
  const __bf16* pk_oWv2 = pack(oWv2, 64, 64);
  const __bf16* pk_oWg  = pack(oWg, 64, 256);
  const __bf16* pk_oWs  = pack(oWs, 256, 320);
  const __bf16* pk_cWv1 = pack(cWv1, 64, 64);
  const __bf16* pk_cWv2 = pack(cWv2, 64, 64);
  const __bf16* pk_cWg  = pack(cWg, 64, 256);
  const __bf16* pk_cWs  = pack(cWs, 256, 320);
  const __bf16* pk_aDir = pack(aDir, 64, 64);
  const __bf16* pk_tWv1 = pack(tWv1, 64, 64);
  const __bf16* pk_tWv2 = pack(tWv2, 64, 64);
  const __bf16* pk_tWg  = pack(tWg, 64, 256);
  const __bf16* pk_tWs  = pack(tWs, 256, 320);

  float* ns   = (float*)alloc_bytes((size_t)N_NODES * 256 * 4);
  float* nv   = (float*)alloc_bytes((size_t)N_NODES * 192 * 4);
  float* cs   = (float*)alloc_bytes((size_t)N_NODES * 256 * 4);
  float* cv   = (float*)alloc_bytes((size_t)N_NODES * 192 * 4);
  float* aggS = (float*)alloc_bytes((size_t)N_NODES * 256 * 4);
  float* aggV = (float*)alloc_bytes((size_t)N_NODES * 192 * 4);

  zero_kernel<<<2048, 256, 0, stream>>>(aggS, (long)N_NODES * 256);
  zero_kernel<<<2048, 256, 0, stream>>>(aggV, (long)N_NODES * 192);

  node_gv_kernel<<<N_NODES / 16, 256, 0, stream>>>(
      x_sca, x_vec,
      pk_nWv1, pk_nWv2, pk_nWg, nGb, pk_nWs,
      pk_cWv1, pk_cWv2, pk_cWg, cGb, pk_cWs,
      ns, nv, cs, cv);

  EdgeArgs ea;
  ea.eidx = eidx; ea.efeat = efeat; ea.evec = evec; ea.vee_w = vee_w;
  ea.eWv1 = pk_eWv1; ea.eWv2 = pk_eWv2; ea.eWg = pk_eWg; ea.eWs = pk_eWs; ea.eGb = egGb;
  ea.eDir = pk_eDir;
  ea.scaW = pk_scaW; ea.scaB = scaB;
  ea.e2nW = pk_e2nW; ea.e2nB = e2nB;
  ea.n2eW = pk_n2eW; ea.n2eB = n2eB;
  ea.WevW = pk_WevW;
  ea.oWv1 = pk_oWv1; ea.oWv2 = pk_oWv2; ea.oWg = pk_oWg; ea.oWs = pk_oWs; ea.oGb = oGb;
  ea.ns = ns; ea.nv = nv; ea.aggS = aggS; ea.aggV = aggV;
  edge_message_kernel<<<N_EDGES / 16, 256, 0, stream>>>(ea);

  OutArgs oa;
  oa.cs = cs; oa.cv = cv; oa.aggS = aggS; oa.aggV = aggV;
  oa.lnsg = lnsg; oa.lnsb = lnsb; oa.lnvg = lnvg; oa.lnvb = lnvb;
  oa.aDir = pk_aDir;
  oa.tWv1 = pk_tWv1; oa.tWv2 = pk_tWv2; oa.tWg = pk_tWg; oa.tWs = pk_tWs; oa.tGb = tGb;
  oa.out = (float*)d_out;
  node_out_kernel<<<N_NODES / 16, 256, 0, stream>>>(oa);
}
